// MCAF_30073361007275
// MI455X (gfx1250) — matrix operand order repaired and checked by reading, not compiled, not run on hardware
//
#include <hip/hip_runtime.h>
#include <hip/hip_bf16.h>
#include <math.h>

// CDNA5 / gfx1250: wave32, WMMA 16x16x32 f16 -> f32.
typedef __attribute__((ext_vector_type(16))) _Float16 v16h;
typedef __attribute__((ext_vector_type(8)))  _Float16 v8h;
typedef __attribute__((ext_vector_type(8)))  float    v8f;

#define LDA 336   // A-staging row stride in halfs (>=320, 16B-aligned rows)

// ---------------- workspace layout (element = _Float16) ----------------
constexpr int OFF_WV  = 0;                     // eeg wv   [64][64]   (62x62 padded)
constexpr int OFF_OW  = OFF_WV  + 64*64;       // eeg ow   [64][64]
constexpr int OFF_CW  = OFF_OW  + 64*64;       // eeg conv [32][320]  (K=310 padded)
constexpr int OFF_FWE = OFF_CW  + 32*320;      // eeg fw   [64][32]
constexpr int OFF_FWO = OFF_FWE + 64*32;       // eog fw   [64][32]
constexpr int OFF_FUS = OFF_FWO + 64*32;       // fus_w    [64][128]
constexpr int OFF_WVT = OFF_FUS + 64*128;      // 2 x value proj [64][64]
constexpr int OFF_OWT = OFF_WVT + 2*64*64;     // 2 x out   proj [64][64]
constexpr int OFF_W1  = OFF_OWT + 2*64*64;     // 2 x mlp w1 [256][64]
constexpr int OFF_W2  = OFF_W1  + 2*256*64;    // 2 x mlp w2 [64][256]
constexpr int WS_HALF = OFF_W2  + 2*64*256;    // ~225 KB

// ---------------- weight convert/pad kernel ----------------
__global__ void cvt_pad_f16(const float* __restrict__ src, _Float16* __restrict__ dst,
                            int N, int K, int Npad, int Kpad) {
    int i = blockIdx.x * 256 + threadIdx.x;
    if (i >= Npad * Kpad) return;
    int n = i / Kpad, k = i - n * Kpad;
    float v = (n < N && k < K) ? src[n * K + k] : 0.f;
    dst[i] = (_Float16)v;
}

// ---------------- WMMA helpers ----------------
__device__ __forceinline__ v8f wmma16(v16h a, v16h b, v8f c) {
    return __builtin_amdgcn_wmma_f32_16x16x32_f16(false, a, false, b, (short)0, c, false, false);
}

// A-fragment (16x32 f16) from LDS staging tile (row-major [16][LDA]).
// lane<16 holds row m=lane, K = k0+0..15;
// lane>=16 holds row m=lane-16, K = k0+16..31 (contiguous).
__device__ __forceinline__ v16h load_afrag(const _Float16* Abase, int k0, int lane) {
    int m = lane & 15, g = lane >> 4;
    const _Float16* p = Abase + m * LDA + k0 + g * 16;
    v8h lo = *(const v8h*)p;
    v8h hi = *(const v8h*)(p + 8);
    v16h r;
#pragma unroll
    for (int i = 0; i < 8; ++i) { r[i] = lo[i]; r[i + 8] = hi[i]; }
    return r;
}

// B-fragment (32x16 f16), B[k][n] = W[n0+n][k0+k] for out = act @ W^T.
// lane<16: n=lane, K=k0+0..15; lane>=16: n=lane-16, K=k0+16..31 (contiguous).
__device__ __forceinline__ v16h load_bfrag(const _Float16* __restrict__ Wp, int Kpad,
                                           int n0, int k0, int lane) {
    int g = lane >> 4;
    const _Float16* p = Wp + (size_t)(n0 + (lane & 15)) * Kpad + k0 + g * 16;
    v8h lo = *(const v8h*)p;
    v8h hi = *(const v8h*)(p + 8);
    v16h r;
#pragma unroll
    for (int i = 0; i < 8; ++i) { r[i] = lo[i]; r[i + 8] = hi[i]; }
    return r;
}

// 16 rows x 64 cols output (4 n-tiles), K = 32*KS starting at A col aK0.
template <int KS>
__device__ __forceinline__ void gemm4(const _Float16* Abase, int aK0,
                                      const _Float16* __restrict__ Wp, int Kpad,
                                      int lane, v8f acc[4]) {
    v16h af[KS];
#pragma unroll
    for (int s = 0; s < KS; ++s) af[s] = load_afrag(Abase, aK0 + 32 * s, lane);
#pragma unroll
    for (int t = 0; t < 4; ++t) {
        v8f c = {};
#pragma unroll
        for (int s = 0; s < KS; ++s)
            c = wmma16(af[s], load_bfrag(Wp, Kpad, 16 * t, 32 * s, lane), c);
        acc[t] = c;
    }
}

struct Params {
    const float *eeg, *eog;
    const float *eeg_inb, *eeg_ob, *eeg_cb, *eeg_fb;
    const float *eog_inw, *eog_inb, *eog_ow, *eog_ob, *eog_cw, *eog_cb, *eog_fb;
    const float *fus_b;
    const float *ln1s, *ln1b, *tinb, *tob, *ln2s, *ln2b, *b1, *b2;
    const float *fns, *fnb, *clsw, *clsb;
    const _Float16 *W;
    float *out;
};

// 2 waves/block, 16 batch rows per wave.
__global__ __launch_bounds__(64) void mcaf_main(Params p) {
    __shared__ __align__(16) _Float16 sA[2][16][LDA];  // f16 A staging (21.5 KB)
    __shared__ float sF[2][16][64];                     // f32 feat/residual
    __shared__ float sO[2][16][64];                     // obar (eeg channel gains)
    __shared__ float sOe[2][16];                        // obar (eog scalar)
    __shared__ float sMu[2][16], sRs[2][16];

    const int w    = threadIdx.x >> 5;
    const int lane = threadIdx.x & 31;
    const int r0   = blockIdx.x * 32 + w * 16;

    _Float16* A = &sA[w][0][0];
    float*    F = &sF[w][0][0];
    float*    O = &sO[w][0][0];

    const int nn = lane & 15;
    const int mb = (lane >> 4) * 8;

    // ---- stage 1a: pooled = mean_L(eeg) -> A[:, 0:64] (cols 62,63 zero) ----
    for (int idx = lane; idx < 16 * 64; idx += 32) {
        int m = idx >> 6, c = idx & 63;
        float v = 0.f;
        if (c < 62) {
            const float* x = p.eeg + ((size_t)(r0 + m) * 62 + c) * 5;
            v = (x[0] + x[1] + x[2] + x[3] + x[4]) * 0.2f;
        }
        A[m * LDA + c] = (_Float16)v;
    }
    __syncthreads();

    // ---- stage 1b: vbar = pooled @ Wv^T + bv ----
    {
        v8f acc[4];
        gemm4<2>(A, 0, p.W + OFF_WV, 64, lane, acc);
        __syncthreads();
#pragma unroll
        for (int t = 0; t < 4; ++t) {
            int n = 16 * t + nn;
            float bv = (n < 62) ? p.eeg_inb[124 + n] : 0.f;
#pragma unroll
            for (int g = 0; g < 8; ++g)
                A[(mb + g) * LDA + n] = (_Float16)((n < 62) ? (acc[t][g] + bv) : 0.f);
        }
    }
    __syncthreads();

    // ---- stage 1c: obar = vbar @ Ow^T + ob -> sO (f32) ----
    {
        v8f acc[4];
        gemm4<2>(A, 0, p.W + OFF_OW, 64, lane, acc);
#pragma unroll
        for (int t = 0; t < 4; ++t) {
            int n = 16 * t + nn;
            float ob = (n < 62) ? p.eeg_ob[n] : 0.f;
#pragma unroll
            for (int g = 0; g < 8; ++g)
                O[(mb + g) * 64 + n] = (n < 62) ? (acc[t][g] + ob) : 0.f;
        }
    }
    __syncthreads();

    // ---- stage 1d: conv A[b, c*5+l] = x[b,c,l] * obar[b,c]  (K pad 310->320) ----
    for (int idx = lane; idx < 16 * 320; idx += 32) {
        int m = idx / 320, k = idx - m * 320;
        _Float16 v = (_Float16)0.f;
        if (k < 310) {
            int c = k / 5, l = k - 5 * c;
            v = (_Float16)(p.eeg[((size_t)(r0 + m) * 62 + c) * 5 + l] * O[m * 64 + c]);
        }
        A[m * LDA + k] = v;
    }
    __syncthreads();

    // ---- stage 1e: h_eeg = convA @ CW^T  (N=32, K=320) ----
    v8f hc0 = {}, hc1 = {};
#pragma unroll
    for (int s = 0; s < 10; ++s) {
        v16h a = load_afrag(A, 32 * s, lane);
        hc0 = wmma16(a, load_bfrag(p.W + OFF_CW, 320, 0, 32 * s, lane), hc0);
        hc1 = wmma16(a, load_bfrag(p.W + OFF_CW, 320, 16, 32 * s, lane), hc1);
    }
    // eog channel gain (softmax collapses -> scalar per row)
    if (lane < 16) {
        const float* xe = p.eog + (size_t)(r0 + lane) * 33;
        float s = 0.f;
        for (int l = 0; l < 33; ++l) s += xe[l];
        s *= (1.f / 33.f);
        float vb = s * p.eog_inw[2] + p.eog_inb[2];
        sOe[w][lane] = vb * p.eog_ow[0] + p.eog_ob[0];
    }
    __syncthreads();
    // scatter elu(h_eeg + cb) -> A[:, 160:192]
#pragma unroll
    for (int g = 0; g < 8; ++g) {
        int m = mb + g;
        float a0 = hc0[g] + p.eeg_cb[nn];
        float a1 = hc1[g] + p.eeg_cb[16 + nn];
        A[m * LDA + 160 + nn]      = (_Float16)(a0 > 0.f ? a0 : expm1f(a0));
        A[m * LDA + 176 + nn]      = (_Float16)(a1 > 0.f ? a1 : expm1f(a1));
    }
    // h_eog (VALU, tiny: 32x33) -> elu -> A[:, 192:224]
    for (int idx = lane; idx < 16 * 32; idx += 32) {
        int m = idx >> 5, o = idx & 31;
        const float* xe  = p.eog + (size_t)(r0 + m) * 33;
        const float* cwo = p.eog_cw + o * 33;
        float d = 0.f;
        for (int l = 0; l < 33; ++l) d += xe[l] * cwo[l];
        float h = sOe[w][m] * d + p.eog_cb[o];
        A[m * LDA + 192 + o] = (_Float16)(h > 0.f ? h : expm1f(h));
    }
    __syncthreads();

    // ---- ef = h_eeg @ fw^T, of = h_eog @ fw^T (K=32 each) ----
    v8f aE[4], aO4[4];
    gemm4<1>(A, 160, p.W + OFF_FWE, 32, lane, aE);
    gemm4<1>(A, 192, p.W + OFF_FWO, 32, lane, aO4);
    __syncthreads();
#pragma unroll
    for (int t = 0; t < 4; ++t) {
        int n = 16 * t + nn;
#pragma unroll
        for (int g = 0; g < 8; ++g) {
            A[(mb + g) * LDA + n]      = (_Float16)(aE[t][g] + p.eeg_fb[n]);
            A[(mb + g) * LDA + 64 + n] = (_Float16)(aO4[t][g] + p.eog_fb[n]);
        }
    }
    __syncthreads();

    // ---- feat = [ef|of] @ fus^T + fus_b + pe0  (K=128) -> sF f32 ----
    {
        v8f acc[4];
        gemm4<4>(A, 0, p.W + OFF_FUS, 128, lane, acc);
#pragma unroll
        for (int t = 0; t < 4; ++t) {
            int n = 16 * t + nn;
            float add = p.fus_b[n] + (float)(n & 1);  // pe at pos 0
#pragma unroll
            for (int g = 0; g < 8; ++g) F[(mb + g) * 64 + n] = acc[t][g] + add;
        }
    }
    __syncthreads();

    // LN(F) -> f16 A[:, 0:64]
    auto layernorm_to_A = [&](const float* s_, const float* b_) {
        if (lane < 16) {
            float mu = 0.f;
#pragma unroll
            for (int c = 0; c < 64; ++c) mu += F[lane * 64 + c];
            mu *= (1.f / 64.f);
            float var = 0.f;
#pragma unroll
            for (int c = 0; c < 64; ++c) { float d = F[lane * 64 + c] - mu; var += d * d; }
            var *= (1.f / 64.f);
            sMu[w][lane] = mu;
            sRs[w][lane] = rsqrtf(var + 1e-5f);
        }
        __syncthreads();
        for (int idx = lane; idx < 16 * 64; idx += 32) {
            int m = idx >> 6, c = idx & 63;
            A[m * LDA + c] = (_Float16)((F[m * 64 + c] - sMu[w][m]) * sRs[w][m] * s_[c] + b_[c]);
        }
        __syncthreads();
    };

    // ---- 2 transformer layers (seq_len 1: attn == Wo(Wv h)+biases) ----
    for (int i = 0; i < 2; ++i) {
        const _Float16* WV  = p.W + OFF_WVT + i * 64 * 64;
        const _Float16* OWt = p.W + OFF_OWT + i * 64 * 64;
        const _Float16* W1  = p.W + OFF_W1  + i * 256 * 64;
        const _Float16* W2  = p.W + OFF_W2  + i * 64 * 256;

        layernorm_to_A(p.ln1s + i * 64, p.ln1b + i * 64);
        {   // v = h @ Wv^T + bv  -> A[:,0:64]
            v8f acc[4];
            gemm4<2>(A, 0, WV, 64, lane, acc);
            __syncthreads();
#pragma unroll
            for (int t = 0; t < 4; ++t) {
                int n = 16 * t + nn;
                float bv = p.tinb[i * 192 + 128 + n];
#pragma unroll
                for (int g = 0; g < 8; ++g)
                    A[(mb + g) * LDA + n] = (_Float16)(acc[t][g] + bv);
            }
            __syncthreads();
        }
        {   // F += v @ Ow^T + ob
            v8f acc[4];
            gemm4<2>(A, 0, OWt, 64, lane, acc);
#pragma unroll
            for (int t = 0; t < 4; ++t) {
                int n = 16 * t + nn;
                float ob = p.tob[i * 64 + n];
#pragma unroll
                for (int g = 0; g < 8; ++g) F[(mb + g) * 64 + n] += acc[t][g] + ob;
            }
            __syncthreads();
        }
        layernorm_to_A(p.ln2s + i * 64, p.ln2b + i * 64);
        {   // MLP: G = gelu(h2 @ W1^T + b1) -> A[:, 64:320]; F += G @ W2^T + b2
            v16h a0 = load_afrag(A, 0, lane), a1 = load_afrag(A, 32, lane);
#pragma unroll
            for (int jt = 0; jt < 16; ++jt) {
                v8f acc = {};
                acc = wmma16(a0, load_bfrag(W1, 64, 16 * jt, 0, lane), acc);
                acc = wmma16(a1, load_bfrag(W1, 64, 16 * jt, 32, lane), acc);
                int n = 16 * jt + nn;
                float b1v = p.b1[i * 256 + n];
#pragma unroll
                for (int g = 0; g < 8; ++g) {
                    float x = acc[g] + b1v;
                    float ge = 0.5f * x * (1.f + erff(x * 0.70710678118654752f));
                    A[(mb + g) * LDA + 64 + n] = (_Float16)ge;
                }
            }
            __syncthreads();
            v8f acc[4];
#pragma unroll
            for (int t = 0; t < 4; ++t) { v8f z = {}; acc[t] = z; }
#pragma unroll
            for (int s = 0; s < 8; ++s) {
                v16h a = load_afrag(A, 64 + 32 * s, lane);
#pragma unroll
                for (int t = 0; t < 4; ++t)
                    acc[t] = wmma16(a, load_bfrag(W2, 256, 16 * t, 32 * s, lane), acc[t]);
            }
#pragma unroll
            for (int t = 0; t < 4; ++t) {
                int n = 16 * t + nn;
                float b2v = p.b2[i * 64 + n];
#pragma unroll
                for (int g = 0; g < 8; ++g) F[(mb + g) * 64 + n] += acc[t][g] + b2v;
            }
            __syncthreads();
        }
    }

    // ---- final LN + classifier (N=3: VALU) ----
    if (lane < 16) {
        float mu = 0.f;
#pragma unroll
        for (int c = 0; c < 64; ++c) mu += F[lane * 64 + c];
        mu *= (1.f / 64.f);
        float var = 0.f;
#pragma unroll
        for (int c = 0; c < 64; ++c) { float d = F[lane * 64 + c] - mu; var += d * d; }
        float rs = rsqrtf(var * (1.f / 64.f) + 1e-5f);
        float o0 = p.clsb[0], o1 = p.clsb[1], o2 = p.clsb[2];
#pragma unroll
        for (int c = 0; c < 64; ++c) {
            float hn = (F[lane * 64 + c] - mu) * rs * p.fns[c] + p.fnb[c];
            o0 += hn * p.clsw[c];
            o1 += hn * p.clsw[64 + c];
            o2 += hn * p.clsw[128 + c];
        }
        size_t b = (size_t)(r0 + lane) * 3;
        p.out[b + 0] = o0; p.out[b + 1] = o1; p.out[b + 2] = o2;
    }
}

extern "C" void kernel_launch(void* const* d_in, const int* in_sizes, int n_in,
                              void* d_out, int out_size, void* d_ws, size_t ws_size,
                              hipStream_t stream) {
    const float* eeg     = (const float*)d_in[0];
    const float* eog     = (const float*)d_in[1];
    const float* eeg_inw = (const float*)d_in[2];
    const float* eeg_inb = (const float*)d_in[3];
    const float* eeg_ow  = (const float*)d_in[4];
    const float* eeg_ob  = (const float*)d_in[5];
    const float* eog_inw = (const float*)d_in[6];
    const float* eog_inb = (const float*)d_in[7];
    const float* eog_ow  = (const float*)d_in[8];
    const float* eog_ob  = (const float*)d_in[9];
    const float* eeg_cw  = (const float*)d_in[10];
    const float* eeg_cb  = (const float*)d_in[11];
    const float* eeg_fw  = (const float*)d_in[12];
    const float* eeg_fb  = (const float*)d_in[13];
    const float* eog_cw  = (const float*)d_in[14];
    const float* eog_cb  = (const float*)d_in[15];
    const float* eog_fw  = (const float*)d_in[16];
    const float* eog_fb  = (const float*)d_in[17];
    const float* fus_w   = (const float*)d_in[18];
    const float* fus_b   = (const float*)d_in[19];
    const float* ln1s    = (const float*)d_in[20];
    const float* ln1b    = (const float*)d_in[21];
    const float* tl_inw  = (const float*)d_in[22];
    const float* tl_inb  = (const float*)d_in[23];
    const float* tl_ow   = (const float*)d_in[24];
    const float* tl_ob   = (const float*)d_in[25];
    const float* ln2s    = (const float*)d_in[26];
    const float* ln2b    = (const float*)d_in[27];
    const float* tl_w1   = (const float*)d_in[28];
    const float* tl_b1   = (const float*)d_in[29];
    const float* tl_w2   = (const float*)d_in[30];
    const float* tl_b2   = (const float*)d_in[31];
    const float* fns     = (const float*)d_in[32];
    const float* fnb     = (const float*)d_in[33];
    const float* clsw    = (const float*)d_in[34];
    const float* clsb    = (const float*)d_in[35];

    _Float16* W = (_Float16*)d_ws;
    auto cvt = [&](const float* src, int off, int N, int K, int Npad, int Kpad) {
        int total = Npad * Kpad;
        cvt_pad_f16<<<(total + 255) / 256, 256, 0, stream>>>(src, W + off, N, K, Npad, Kpad);
    };
    cvt(eeg_inw + 124 * 62, OFF_WV, 62, 62, 64, 64);     // value proj (wv slice)
    cvt(eeg_ow,             OFF_OW, 62, 62, 64, 64);
    cvt(eeg_cw,             OFF_CW, 32, 310, 32, 320);
    cvt(eeg_fw,             OFF_FWE, 64, 32, 64, 32);
    cvt(eog_fw,             OFF_FWO, 64, 32, 64, 32);
    cvt(fus_w,              OFF_FUS, 64, 128, 64, 128);
    for (int i = 0; i < 2; ++i) {
        cvt(tl_inw + (i * 192 + 128) * 64, OFF_WVT + i * 64 * 64, 64, 64, 64, 64);
        cvt(tl_ow + i * 64 * 64,           OFF_OWT + i * 64 * 64, 64, 64, 64, 64);
        cvt(tl_w1 + i * 256 * 64,          OFF_W1 + i * 256 * 64, 256, 64, 256, 64);
        cvt(tl_w2 + i * 64 * 256,          OFF_W2 + i * 64 * 256, 64, 256, 64, 256);
    }

    int B = in_sizes[0] / 310;  // 131072
    Params p;
    p.eeg = eeg; p.eog = eog;
    p.eeg_inb = eeg_inb; p.eeg_ob = eeg_ob; p.eeg_cb = eeg_cb; p.eeg_fb = eeg_fb;
    p.eog_inw = eog_inw; p.eog_inb = eog_inb; p.eog_ow = eog_ow; p.eog_ob = eog_ob;
    p.eog_cw = eog_cw; p.eog_cb = eog_cb; p.eog_fb = eog_fb;
    p.fus_b = fus_b;
    p.ln1s = ln1s; p.ln1b = ln1b; p.tinb = tl_inb; p.tob = tl_ob;
    p.ln2s = ln2s; p.ln2b = ln2b; p.b1 = tl_b1; p.b2 = tl_b2;
    p.fns = fns; p.fnb = fnb; p.clsw = clsw; p.clsb = clsb;
    p.W = W; p.out = (float*)d_out;

    mcaf_main<<<dim3(B / 32), dim3(64), 0, stream>>>(p);
}
